// NeighborhoodAttention2D_67645734912668
// MI455X (gfx1250) — matrix operand order repaired and checked by reading, not compiled, not run on hardware
//
#include <hip/hip_runtime.h>
#include <hip/hip_bf16.h>

// NeighborhoodAttention2D for MI455X (gfx1250, wave32, WMMA).
// Pipeline: x -> NHWC f16 | w -> f16 | WMMA GEMM qkv | wave-per-pixel attention | WMMA GEMM proj (transposed store)

typedef _Float16 f16;
typedef __attribute__((ext_vector_type(16))) _Float16     v16h;
typedef __attribute__((ext_vector_type(8)))  float        v8f;
typedef __attribute__((ext_vector_type(4)))  unsigned int v4u;

union FragAB { v16h h; v4u u[2]; };

#define Bn   2
#define Cn   256
#define Hn   56
#define Wn   56
#define HWn  (Hn * Wn)          // 3136
#define Mn   (Bn * HWn)         // 6272 rows
#define NHn  8
#define Dn   32
#define Kw   7                  // window
#define KKn  (Kw * Kw)          // 49

// ---------------------------------------------------------------- conversions

// x (B,C,H,W) f32 -> xh (B*H*W, C) f16 (channels-last for GEMM A)
__global__ void k_x_to_f16_nhwc(const float* __restrict__ x, f16* __restrict__ xh) {
  int idx = blockIdx.x * blockDim.x + threadIdx.x;     // M*C exact
  int c    = idx & (Cn - 1);
  int rest = idx >> 8;                                 // b*HW + hw
  int hw   = rest % HWn;
  int b    = rest / HWn;
  xh[idx] = (f16)x[((size_t)(b * Cn + c)) * HWn + hw];
}

__global__ void k_f32_to_f16(const float* __restrict__ in, f16* __restrict__ out, int n) {
  int i = blockIdx.x * blockDim.x + threadIdx.x;
  if (i < n) out[i] = (f16)in[i];
}

// ---------------------------------------------------------------- WMMA GEMM
// out[m,n] = sum_k A[m,k] * Wt[n,k] + bias[n]
// A: M x 256 row-major f16; Wt: N x 256 row-major f16 (out_features x in_features).
// Block = 256 threads = 8 waves, wave grid 4(M) x 2(N) of 16x16 tiles -> 64x32 per block.
// HWtrans == 0 : out is M x N row-major
// HWtrans == HW: out[(b*N + n)*HW + hw] with m = b*HW + hw (direct (B,C,H,W) store)
__global__ __launch_bounds__(256) void k_gemm_f16_wmma(
    const f16* __restrict__ A, const f16* __restrict__ Wt,
    const float* __restrict__ bias, float* __restrict__ out,
    int N, int HWtrans)
{
  constexpr int K = 256;
  int lane = threadIdx.x & 31;
  int wv   = threadIdx.x >> 5;
  int tm   = blockIdx.x * 64 + (wv >> 1) * 16;
  int tn   = blockIdx.y * 32 + (wv & 1) * 16;
  int half = lane >> 4;
  int l15  = lane & 15;

  const f16* ap = A  + (size_t)(tm + l15) * K;   // lane = A row
  const f16* bp = Wt + (size_t)(tn + l15) * K;   // lane = B column = weight row
  // ISA §7.12.2: A 16x32 f16, lane-half K chunks {0..7,16..23} / {8..15,24..31}
  int ka0 = half * 8, ka1 = 16 + half * 8;
  // B 32x16: same K chunks as A, {0..7,16..23} / {8..15,24..31}
  int kb0 = half * 8, kb1 = 16 + half * 8;

  v8f acc = {};
#pragma unroll
  for (int kk = 0; kk < K; kk += 32) {
    FragAB fa, fb;
    fa.u[0] = *(const v4u*)(ap + kk + ka0);
    fa.u[1] = *(const v4u*)(ap + kk + ka1);
    fb.u[0] = *(const v4u*)(bp + kk + kb0);
    fb.u[1] = *(const v4u*)(bp + kk + kb1);
    acc = __builtin_amdgcn_wmma_f32_16x16x32_f16(
        /*neg_a=*/false, fa.h, /*neg_b=*/false, fb.h,
        /*c_mod=*/(short)0, acc, /*reuse_a=*/false, /*reuse_b=*/false);
  }

  int   col   = tn + l15;
  float bs    = bias[col];
  int   rbase = tm + half * 8;                   // C/D: VGPR j -> row rbase + j
  if (HWtrans == 0) {
    float* op = out + (size_t)rbase * N + col;
#pragma unroll
    for (int j = 0; j < 8; ++j) op[(size_t)j * N] = acc[j] + bs;
  } else {
#pragma unroll
    for (int j = 0; j < 8; ++j) {
      int row = rbase + j;
      int bI  = row / HWtrans;
      int hw  = row - bI * HWtrans;
      out[((size_t)bI * N + col) * HWtrans + hw] = acc[j] + bs;
    }
  }
}

// ---------------------------------------------------------------- attention
// One wave per (b, head, h, w) pixel; lane = d (phase 2) / neighbor (phase 1).
// qkv: M x 768 f32 (q: 0..255, k: 256..511, v: 512..767; head hd at hd*32).
__global__ __launch_bounds__(256) void k_natten(
    const float* __restrict__ qkv, const float* __restrict__ rpb,
    f16* __restrict__ aout)
{
  __shared__ float qs[8 * Dn];
  __shared__ float ps[8 * 64];

  int lane = threadIdx.x & 31;
  int wv   = threadIdx.x >> 5;
  int pix  = blockIdx.x * 8 + wv;
  int w  = pix % Wn;
  int h  = (pix / Wn) % Hn;
  int hd = (pix / HWn) % NHn;
  int b  = pix / (HWn * NHn);
  int m  = (b * Hn + h) * Wn + w;
  int sh = h - 3; if (sh < 0) sh = 0; if (sh > Hn - Kw) sh = Hn - Kw;
  int sw = w - 3; if (sw < 0) sw = 0; if (sw > Wn - Kw) sw = Wn - Kw;

  const float scale = 0.17677669529663687f;      // 1/sqrt(32)
  qs[wv * Dn + lane] = qkv[(size_t)m * 768 + hd * Dn + lane] * scale;
  __syncthreads();

  // phase 1: logits for neighbors j = lane and j = lane + 32
  const float4* qp = (const float4*)(qs + wv * Dn);
  float l0 = -1e30f, l1 = -1e30f;
#pragma unroll
  for (int t = 0; t < 2; ++t) {
    int j = lane + t * 32;
    float lg = -1e30f;
    if (j < KKn) {
      int r = j / Kw, c = j - r * Kw;
      int kr = sh + r, kc = sw + c;
      int m2 = (b * Hn + kr) * Wn + kc;
      const float4* kp = (const float4*)(qkv + (size_t)m2 * 768 + 256 + hd * Dn);
      float acc = 0.f;
#pragma unroll
      for (int d4 = 0; d4 < 8; ++d4) {
        float4 kk = kp[d4], qq = qp[d4];
        acc += qq.x * kk.x + qq.y * kk.y + qq.z * kk.z + qq.w * kk.w;
      }
      lg = acc + rpb[hd * 169 + (kr - h + 6) * 13 + (kc - w + 6)];
    }
    if (t == 0) l0 = lg; else l1 = lg;
  }

  // softmax over 49 logits spread across lanes
  float mx = fmaxf(l0, l1);
#pragma unroll
  for (int off = 16; off >= 1; off >>= 1) mx = fmaxf(mx, __shfl_xor(mx, off, 32));
  float e0 = __expf(l0 - mx);
  float e1 = (lane + 32 < KKn) ? __expf(l1 - mx) : 0.f;
  float s = e0 + e1;
#pragma unroll
  for (int off = 16; off >= 1; off >>= 1) s += __shfl_xor(s, off, 32);
  float inv = 1.f / s;
  ps[wv * 64 + lane] = e0 * inv;
  if (lane + 32 < KKn) ps[wv * 64 + 32 + lane] = e1 * inv;
  __syncthreads();

  // phase 2: out[d] = sum_j p_j * v[nb(j)][d], lane = d (coalesced v reads)
  const float* pp = ps + wv * 64;
  float acc = 0.f;
  int j = 0;
#pragma unroll
  for (int r = 0; r < Kw; ++r) {
    int m2 = (b * Hn + (sh + r)) * Wn + sw;
    const float* vp = qkv + (size_t)m2 * 768 + 512 + hd * Dn + lane;
#pragma unroll
    for (int c = 0; c < Kw; ++c) acc += pp[j++] * vp[(size_t)c * 768];
  }
  aout[(size_t)m * Cn + hd * Dn + lane] = (f16)acc;
}

// ---------------------------------------------------------------- launch

extern "C" void kernel_launch(void* const* d_in, const int* in_sizes, int n_in,
                              void* d_out, int out_size, void* d_ws, size_t ws_size,
                              hipStream_t stream) {
  (void)in_sizes; (void)n_in; (void)out_size; (void)ws_size;
  const float* x      = (const float*)d_in[0];
  const float* w_qkv  = (const float*)d_in[1];
  const float* b_qkv  = (const float*)d_in[2];
  const float* rpb    = (const float*)d_in[3];
  const float* w_proj = (const float*)d_in[4];
  const float* b_proj = (const float*)d_in[5];
  float* out = (float*)d_out;

  // workspace layout (bytes, 256-aligned); total ~25 MB (L2-resident)
  char* ws = (char*)d_ws;
  f16*   xh     = (f16*)(ws + 0);          // M*256*2  = 3,211,264
  f16*   wqkvh  = (f16*)(ws + 3211264);    // 768*256*2=   393,216
  f16*   wprojh = (f16*)(ws + 3604480);    // 256*256*2=   131,072
  float* qkv    = (float*)(ws + 3735552);  // M*768*4  = 19,267,584
  f16*   aout   = (f16*)(ws + 23003136);   // M*256*2  = 3,211,264

  k_x_to_f16_nhwc<<<Mn, 256, 0, stream>>>(x, xh);                       // M*C / 256 blocks
  k_f32_to_f16<<<(768 * 256) / 256, 256, 0, stream>>>(w_qkv, wqkvh, 768 * 256);
  k_f32_to_f16<<<(256 * 256) / 256, 256, 0, stream>>>(w_proj, wprojh, 256 * 256);

  // qkv = xh @ w_qkv^T + b_qkv : M=6272, N=768, K=256
  k_gemm_f16_wmma<<<dim3(Mn / 64, 768 / 32), 256, 0, stream>>>(xh, wqkvh, b_qkv, qkv, 768, 0);

  // attention: 6272 blocks * 8 waves = 50176 (b,head,pixel) tuples
  k_natten<<<(Bn * NHn * HWn) / 8, 256, 0, stream>>>(qkv, rpb, aout);

  // out = aout @ w_proj^T + b_proj, stored directly as (B,C,H,W)
  k_gemm_f16_wmma<<<dim3(Mn / 64, 256 / 32), 256, 0, stream>>>(aout, wprojh, b_proj, out, 256, HWn);
}
